// Net_41575283425667
// MI455X (gfx1250) — matrix operand order repaired and checked by reading, not compiled, not run on hardware
//
#include <hip/hip_runtime.h>

#define NUSER 100000
#define NITEM 100000
#define NEDGE 800000
#define NLBL  200000
#define FDIM  256
#define HDIM  512
#define ODIM  256

typedef __attribute__((ext_vector_type(16))) __bf16 v16bf;
typedef __attribute__((ext_vector_type(8)))  float  v8f;

union Frag { unsigned long long q[4]; v16bf v; };

__device__ __forceinline__ unsigned short f2bf(float x) {
  unsigned u = __float_as_uint(x);
  u += 0x7fffu + ((u >> 16) & 1u);            // round-to-nearest-even
  return (unsigned short)(u >> 16);
}
__device__ __forceinline__ float bf2f(unsigned h) {
  return __uint_as_float(h << 16);
}

// CDNA5 async global->LDS copy (16B per lane), tracked by ASYNCcnt.
__device__ __forceinline__ void async_cp16(void* lds, const void* gl) {
  unsigned l = (unsigned)(unsigned long long)lds;          // low 32b = LDS offset
  unsigned long long g = (unsigned long long)gl;
  asm volatile("global_load_async_to_lds_b128 %0, %1, off"
               :: "v"(l), "v"(g) : "memory");
}

// ---------------- utility kernels ----------------
__global__ void k_zero(float* __restrict__ p, long long n) {
  long long i = (long long)blockIdx.x * blockDim.x + threadIdx.x;
  if (i < n) p[i] = 0.f;
}

__global__ void k_count(const int* __restrict__ src, const int* __restrict__ dst,
                        unsigned* __restrict__ cu, unsigned* __restrict__ ci, int e) {
  int i = blockIdx.x * blockDim.x + threadIdx.x;
  if (i < e) {
    atomicAdd(&cu[src[i]], 1u);
    atomicAdd(&ci[dst[i]], 1u);
  }
}

__global__ void k_inv(float* __restrict__ buf, int n) {
  int i = blockIdx.x * blockDim.x + threadIdx.x;
  if (i < n) {
    unsigned c = reinterpret_cast<unsigned*>(buf)[i];
    buf[i] = 1.f / fmaxf((float)c, 1.f);
  }
}

// flat f32 -> bf16 (float4 granularity)
__global__ void k_cvt4(const float* __restrict__ in, unsigned short* __restrict__ out,
                       long long nq) {
  long long i = (long long)blockIdx.x * blockDim.x + threadIdx.x;
  if (i >= nq) return;
  float4 v = ((const float4*)in)[i];
  unsigned lo = (unsigned)f2bf(v.x) | ((unsigned)f2bf(v.y) << 16);
  unsigned hi = (unsigned)f2bf(v.z) | ((unsigned)f2bf(v.w) << 16);
  ((uint2*)out)[i] = make_uint2(lo, hi);
}

// weight transpose-convert: outT[k*N + n] = bf16(in[n*ldin + koff + k]);  N = 1<<nshift
__global__ void k_wcvt(const float* __restrict__ in, int ldin, int koff,
                       unsigned short* __restrict__ outT, int nshift, long long total) {
  long long i = (long long)blockIdx.x * blockDim.x + threadIdx.x;
  if (i >= total) return;
  int n = (int)(i & ((1 << nshift) - 1));
  int k = (int)(i >> nshift);
  outT[i] = f2bf(in[(size_t)n * ldin + koff + k]);
}

// scatter-add: out[si[e]*D + c..c+3] += bf16(feat[gi[e]*D + c..c+3])
__global__ void k_scatter(const unsigned short* __restrict__ feat,
                          const int* __restrict__ gi, const int* __restrict__ si,
                          float* __restrict__ out, long long total, int dshift, int D) {
  long long i = (long long)blockIdx.x * blockDim.x + threadIdx.x;
  if (i >= total) return;
  int e = (int)(i >> dshift);
  int c = (int)((i & ((1LL << dshift) - 1)) << 2);
  const uint2 v = *(const uint2*)&feat[(size_t)gi[e] * D + c];
  float* o = &out[(size_t)si[e] * D + c];
  atomicAdd(o + 0, bf2f(v.x & 0xffffu));
  atomicAdd(o + 1, bf2f(v.x >> 16));
  atomicAdd(o + 2, bf2f(v.y & 0xffffu));
  atomicAdd(o + 3, bf2f(v.y >> 16));
}

// bf16 row gather, D == 256: out[e][:] = table[idx[e]][:]   (8 elems / thread)
__global__ void k_gather(const unsigned short* __restrict__ table,
                         const int* __restrict__ idx,
                         unsigned short* __restrict__ out, long long total) {
  long long i = (long long)blockIdx.x * blockDim.x + threadIdx.x;
  if (i >= total) return;
  int e = (int)(i >> 5);
  int c = (int)((i & 31) << 3);
  *(uint4*)&out[(size_t)e * 256 + c] =
      *(const uint4*)&table[(size_t)idx[e] * 256 + c];
}

// ---------------- WMMA bf16 GEMM: C = epi(A[MxK] @ Wt[KxN]) ----------------
// A bf16 (lda elems), Wt bf16 k-major (ldwt elems). block 256 = 8 wave32,
// tile 128(M) x 64(N), K step 32. Double-buffered async global->LDS staging:
// next K-tile is in flight (ASYNCcnt) while WMMAs consume the current one.
// epilogue (acc != null): v = acc[row,col]*inv[row] + bias[col] + gemm (+relu).
__global__ __launch_bounds__(256)
void k_gemm(const unsigned short* __restrict__ A, int lda,
            const unsigned short* __restrict__ Wt, int ldwt,
            void* __restrict__ Cout, int outBf16,
            int M, int N, int K,
            const float* __restrict__ acc,
            const float* __restrict__ invrow,
            const float* __restrict__ bias,
            int relu) {
  __shared__ __align__(16) unsigned short As[2][128 * 40];  // 128 rows x 32 k (pad 40)
  __shared__ __align__(16) unsigned short Ws[2][32 * 72];   // 32 k x 64 n (pad 72)

  const int tid  = threadIdx.x;
  const int lane = tid & 31;
  const int wv   = tid >> 5;
  const int mblk = blockIdx.x * 128;
  const int n0   = blockIdx.y * 64;

  v8f accv[4] = {};

  const int kb   = (lane >> 4) * 8;           // A-frag K sub-base (ISA layout)
  const int arow = wv * 16 + (lane & 15);     // A-frag local row

  // staging coordinates: 16B (8 bf16) chunks per async op
  const int ar0 = tid >> 2, ac0 = (tid & 3) * 8;   // A rows ar0 / ar0+64
  const int wk  = tid >> 3, wn  = (tid & 7) * 8;   // W chunk
  const int r1 = min(mblk + ar0, M - 1);            // branchless clamp
  const int r2 = min(mblk + ar0 + 64, M - 1);
  const unsigned short* gA1 = &A[(size_t)r1 * lda + ac0];
  const unsigned short* gA2 = &A[(size_t)r2 * lda + ac0];
  const unsigned short* gW  = &Wt[(size_t)wk * ldwt + n0 + wn];

  auto stage = [&](int b, int k0) {
    async_cp16(&As[b][ar0 * 40 + ac0],        gA1 + k0);
    async_cp16(&As[b][(ar0 + 64) * 40 + ac0], gA2 + k0);
    async_cp16(&Ws[b][wk * 72 + wn],          gW + (size_t)k0 * ldwt);
  };

  stage(0, 0);                                // prologue fill
  int buf = 0;
  for (int k0 = 0; k0 < K; k0 += 32) {
    if (k0 + 32 < K) {
      stage(buf ^ 1, k0 + 32);                // prefetch next tile into other buffer
      asm volatile("s_wait_asynccnt 3" ::: "memory");  // current tile (3 older ops) landed
    } else {
      asm volatile("s_wait_asynccnt 0" ::: "memory");
    }
    __syncthreads();

    // A fragment: lanes 0-15 rows M=0..15 (K kb+0..7, kb+16..23); lanes 16-31 kb=8
    Frag a;
    const unsigned short* pa = &As[buf][arow * 40 + kb];
    a.q[0] = *(const unsigned long long*)(pa + 0);
    a.q[1] = *(const unsigned long long*)(pa + 4);
    a.q[2] = *(const unsigned long long*)(pa + 16);
    a.q[3] = *(const unsigned long long*)(pa + 20);

    #pragma unroll
    for (int j = 0; j < 4; ++j) {
      // B fragment: col n = lane&15, packed elems = K kb+0..7, kb+16..23 (same as A)
      Frag b;
      const unsigned short* pb = &Ws[buf][kb * 72 + j * 16 + (lane & 15)];
      #pragma unroll
      for (int t = 0; t < 4; ++t) {
        const int ko = (t & 1) * 4 + (t >> 1) * 16;
        b.q[t] = (unsigned long long)pb[ko * 72] |
                 ((unsigned long long)pb[(ko + 1) * 72] << 16) |
                 ((unsigned long long)pb[(ko + 2) * 72] << 32) |
                 ((unsigned long long)pb[(ko + 3) * 72] << 48);
      }
      accv[j] = __builtin_amdgcn_wmma_f32_16x16x32_bf16(
          false, a.v, false, b.v, (short)0, accv[j], false, false);
    }
    __syncthreads();                          // all waves done reading buf before refill
    buf ^= 1;
  }

  // epilogue: VGPR r, lanes 0-15 -> M=r col=lane; lanes 16-31 -> M=r+8 col=lane-16
  const int half = lane >> 4, coll = lane & 15;
  const int mbase = mblk + wv * 16 + half * 8;
  #pragma unroll
  for (int j = 0; j < 4; ++j) {
    const int col = n0 + j * 16 + coll;
    #pragma unroll
    for (int r = 0; r < 8; ++r) {
      int row = mbase + r;
      if (row < M) {
        float v = accv[j][r];
        if (acc) {
          float s = invrow ? invrow[row] : 1.f;
          v += acc[(size_t)row * N + col] * s + bias[col];
          if (relu) v = fmaxf(v, 0.f);
        }
        if (outBf16) ((unsigned short*)Cout)[(size_t)row * N + col] = f2bf(v);
        else         ((float*)Cout)[(size_t)row * N + col] = v;
      }
    }
  }
}

// ---------------- decoder tail: out = 0.5*dot(h1+h2, w2) + b2 ----------------
__global__ void k_decode(const float* __restrict__ h1, const float* __restrict__ h2,
                         const float* __restrict__ w2, const float* __restrict__ b2,
                         float* __restrict__ out, int ne) {
  int gid = blockIdx.x * blockDim.x + threadIdx.x;
  int e = gid >> 5, lane = gid & 31;
  if (e >= ne) return;
  float p = 0.f;
  #pragma unroll
  for (int j = 0; j < 8; ++j) {
    int c = lane + 32 * j;
    p += (h1[(size_t)e * 256 + c] + h2[(size_t)e * 256 + c]) * w2[c];
  }
  #pragma unroll
  for (int off = 16; off > 0; off >>= 1) p += __shfl_xor(p, off);
  if (lane == 0) out[e] = 0.5f * p + b2[0];
}

// ---------------- driver ----------------
extern "C" void kernel_launch(void* const* d_in, const int* in_sizes, int n_in,
                              void* d_out, int out_size, void* d_ws, size_t ws_size,
                              hipStream_t stream) {
  (void)in_sizes; (void)n_in; (void)out_size; (void)ws_size;
  const float* x_user = (const float*)d_in[0];
  const float* x_item = (const float*)d_in[1];
  const int*   ei     = (const int*)d_in[2];
  const int*   eli    = (const int*)d_in[3];
  const float* c1uiWl = (const float*)d_in[4];
  const float* c1uibl = (const float*)d_in[5];
  const float* c1uiWr = (const float*)d_in[6];
  const float* c1iuWl = (const float*)d_in[7];
  const float* c1iubl = (const float*)d_in[8];
  const float* c1iuWr = (const float*)d_in[9];
  const float* c2uiWl = (const float*)d_in[10];
  const float* c2uibl = (const float*)d_in[11];
  const float* c2uiWr = (const float*)d_in[12];
  const float* c2iuWl = (const float*)d_in[13];
  const float* c2iubl = (const float*)d_in[14];
  const float* c2iuWr = (const float*)d_in[15];
  const float* W1 = (const float*)d_in[16];
  const float* b1 = (const float*)d_in[17];
  const float* W2 = (const float*)d_in[18];
  const float* b2 = (const float*)d_in[19];
  float* out = (float*)d_out;

  const int* src = ei;              // user side
  const int* dst = ei + NEDGE;      // item side
  const int* el0 = eli;
  const int* el1 = eli + NLBL;

  // ---- workspace carve ----
  char* wp = (char*)d_ws;
  auto carve = [&](size_t bytes) {
    char* r = wp; wp += (bytes + 255) & ~(size_t)255; return r;
  };
  const size_t BIGF = (size_t)NITEM * HDIM;            // 51.2M elems
  float* inv_user = (float*)carve((size_t)NUSER * 4);
  float* inv_item = (float*)carve((size_t)NITEM * 4);
  float* FA = (float*)carve(BIGF * 4);                 // acc_item / acc2_item / t1,h1
  float* FB = (float*)carve(BIGF * 4);                 // acc_user / acc2_user / t2,h2
  unsigned short* XU = (unsigned short*)carve((size_t)NUSER * FDIM * 2);
  unsigned short* XI = (unsigned short*)carve((size_t)NITEM * FDIM * 2);
  unsigned short* P0 = (unsigned short*)carve(BIGF * 2);  // hs_u -> h_item -> zu_g
  unsigned short* P1 = (unsigned short*)carve(BIGF * 2);  // hs_i -> h_user -> zi_g
  unsigned short* P2 = (unsigned short*)carve((size_t)NUSER * ODIM * 2); // hs2_u -> z_item
  unsigned short* P3 = (unsigned short*)carve((size_t)NITEM * ODIM * 2); // hs2_i -> z_user
  unsigned short* Wc1uiWl = (unsigned short*)carve((size_t)HDIM * FDIM * 2);
  unsigned short* Wc1iuWl = (unsigned short*)carve((size_t)HDIM * FDIM * 2);
  unsigned short* Wc1uiWr = (unsigned short*)carve((size_t)HDIM * FDIM * 2);
  unsigned short* Wc1iuWr = (unsigned short*)carve((size_t)HDIM * FDIM * 2);
  unsigned short* Wc2uiWl = (unsigned short*)carve((size_t)HDIM * ODIM * 2);
  unsigned short* Wc2iuWl = (unsigned short*)carve((size_t)HDIM * ODIM * 2);
  unsigned short* Wc2uiWr = (unsigned short*)carve((size_t)HDIM * ODIM * 2);
  unsigned short* Wc2iuWr = (unsigned short*)carve((size_t)HDIM * ODIM * 2);
  unsigned short* W1a = (unsigned short*)carve((size_t)ODIM * ODIM * 2);
  unsigned short* W1b = (unsigned short*)carve((size_t)ODIM * ODIM * 2);

  auto nb = [](long long n) { return (unsigned)((n + 255) / 256); };

  // ---- one-shot conversions (bf16 data path) ----
  k_cvt4<<<nb((long long)NUSER * FDIM / 4), 256, 0, stream>>>(x_user, XU, (long long)NUSER * FDIM / 4);
  k_cvt4<<<nb((long long)NITEM * FDIM / 4), 256, 0, stream>>>(x_item, XI, (long long)NITEM * FDIM / 4);
  const long long WC1 = (long long)HDIM * FDIM;   // 512x256
  k_wcvt<<<nb(WC1), 256, 0, stream>>>(c1uiWl, FDIM, 0, Wc1uiWl, 9, WC1);
  k_wcvt<<<nb(WC1), 256, 0, stream>>>(c1iuWl, FDIM, 0, Wc1iuWl, 9, WC1);
  k_wcvt<<<nb(WC1), 256, 0, stream>>>(c1uiWr, FDIM, 0, Wc1uiWr, 9, WC1);
  k_wcvt<<<nb(WC1), 256, 0, stream>>>(c1iuWr, FDIM, 0, Wc1iuWr, 9, WC1);
  const long long WC2 = (long long)ODIM * HDIM;   // 256x512
  k_wcvt<<<nb(WC2), 256, 0, stream>>>(c2uiWl, HDIM, 0, Wc2uiWl, 8, WC2);
  k_wcvt<<<nb(WC2), 256, 0, stream>>>(c2iuWl, HDIM, 0, Wc2iuWl, 8, WC2);
  k_wcvt<<<nb(WC2), 256, 0, stream>>>(c2uiWr, HDIM, 0, Wc2uiWr, 8, WC2);
  k_wcvt<<<nb(WC2), 256, 0, stream>>>(c2iuWr, HDIM, 0, Wc2iuWr, 8, WC2);
  const long long WC3 = (long long)ODIM * ODIM;   // 256x256 halves of W1
  k_wcvt<<<nb(WC3), 256, 0, stream>>>(W1, 2 * ODIM, 0,    W1a, 8, WC3);
  k_wcvt<<<nb(WC3), 256, 0, stream>>>(W1, 2 * ODIM, ODIM, W1b, 8, WC3);

  // ---- degree counts -> inverse means; zero layer-1 accumulators ----
  k_zero<<<nb(NUSER + NITEM), 256, 0, stream>>>(inv_user, NUSER + NITEM);
  k_zero<<<nb(BIGF), 256, 0, stream>>>(FA, BIGF);
  k_zero<<<nb(BIGF), 256, 0, stream>>>(FB, BIGF);
  k_count<<<nb(NEDGE), 256, 0, stream>>>(src, dst, (unsigned*)inv_user, (unsigned*)inv_item, NEDGE);
  k_inv<<<nb(NUSER + NITEM), 256, 0, stream>>>(inv_user, NUSER + NITEM);

  // ---- layer 1 ----
  dim3 g1((NUSER + 127) / 128, HDIM / 64);
  k_gemm<<<g1, 256, 0, stream>>>(XU, FDIM, Wc1uiWl, HDIM, P0, 1, NUSER, HDIM, FDIM,
                                 nullptr, nullptr, nullptr, 0);             // hs_u
  k_gemm<<<g1, 256, 0, stream>>>(XI, FDIM, Wc1iuWl, HDIM, P1, 1, NITEM, HDIM, FDIM,
                                 nullptr, nullptr, nullptr, 0);             // hs_i
  long long t1 = (long long)NEDGE * (HDIM / 4);
  k_scatter<<<nb(t1), 256, 0, stream>>>(P0, src, dst, FA, t1, 7, HDIM);     // -> acc_item
  k_scatter<<<nb(t1), 256, 0, stream>>>(P1, dst, src, FB, t1, 7, HDIM);     // -> acc_user
  k_gemm<<<g1, 256, 0, stream>>>(XI, FDIM, Wc1uiWr, HDIM, P0, 1, NITEM, HDIM, FDIM,
                                 FA, inv_item, c1uibl, 1);                  // h_item -> P0
  k_gemm<<<g1, 256, 0, stream>>>(XU, FDIM, Wc1iuWr, HDIM, P1, 1, NUSER, HDIM, FDIM,
                                 FB, inv_user, c1iubl, 1);                  // h_user -> P1

  // ---- layer 2 ----
  k_zero<<<nb((long long)NITEM * ODIM), 256, 0, stream>>>(FA, (long long)NITEM * ODIM);
  k_zero<<<nb((long long)NUSER * ODIM), 256, 0, stream>>>(FB, (long long)NUSER * ODIM);
  dim3 g2((NUSER + 127) / 128, ODIM / 64);
  k_gemm<<<g2, 256, 0, stream>>>(P1, HDIM, Wc2uiWl, ODIM, P2, 1, NUSER, ODIM, HDIM,
                                 nullptr, nullptr, nullptr, 0);             // hs2_u
  k_gemm<<<g2, 256, 0, stream>>>(P0, HDIM, Wc2iuWl, ODIM, P3, 1, NITEM, ODIM, HDIM,
                                 nullptr, nullptr, nullptr, 0);             // hs2_i
  long long t2 = (long long)NEDGE * (ODIM / 4);
  k_scatter<<<nb(t2), 256, 0, stream>>>(P2, src, dst, FA, t2, 6, ODIM);
  k_scatter<<<nb(t2), 256, 0, stream>>>(P3, dst, src, FB, t2, 6, ODIM);
  k_gemm<<<g2, 256, 0, stream>>>(P0, HDIM, Wc2uiWr, ODIM, P2, 1, NITEM, ODIM, HDIM,
                                 FA, inv_item, c2uibl, 0);                  // z_item -> P2
  k_gemm<<<g2, 256, 0, stream>>>(P1, HDIM, Wc2iuWr, ODIM, P3, 1, NUSER, ODIM, HDIM,
                                 FB, inv_user, c2iubl, 0);                  // z_user -> P3

  // ---- decoder ----
  long long tg = (long long)NLBL * (ODIM / 8);
  k_gather<<<nb(tg), 256, 0, stream>>>(P3, el0, P0, tg);                    // zu -> P0
  k_gather<<<nb(tg), 256, 0, stream>>>(P2, el1, P1, tg);                    // zi -> P1
  dim3 g3((NLBL + 127) / 128, ODIM / 64);
  // W1 = [W1a | W1b]; h1 = relu(zu@W1a^T + zi@W1b^T + b1); h2 swaps zu/zi
  k_gemm<<<g3, 256, 0, stream>>>(P0, ODIM, W1a, ODIM, FA, 0, NLBL, ODIM, ODIM,
                                 nullptr, nullptr, nullptr, 0);             // t1
  k_gemm<<<g3, 256, 0, stream>>>(P1, ODIM, W1b, ODIM, FA, 0, NLBL, ODIM, ODIM,
                                 FA, nullptr, b1, 1);                       // h1 -> FA
  k_gemm<<<g3, 256, 0, stream>>>(P1, ODIM, W1a, ODIM, FB, 0, NLBL, ODIM, ODIM,
                                 nullptr, nullptr, nullptr, 0);             // t2
  k_gemm<<<g3, 256, 0, stream>>>(P0, ODIM, W1b, ODIM, FB, 0, NLBL, ODIM, ODIM,
                                 FB, nullptr, b1, 1);                       // h2 -> FB

  k_decode<<<nb((long long)NLBL * 32), 256, 0, stream>>>(FA, FB, W2, b2, out, NLBL);
}
